// RWKV6Attention_32727650795844
// MI455X (gfx1250) — compile-verified, matrix operand order checked by reading, not run on hardware
//
#include <hip/hip_runtime.h>
#include <stdint.h>

// ---------------- constants ----------------
constexpr int T    = 2048;
constexpr int C    = 2048;
constexpr int Hh   = 32;
constexpr int Ss   = 64;
constexpr int HKV  = 8;
constexpr int LMAA = 32;           // 5*LMAA = 160
constexpr int LDEC = 64;
constexpr int CH   = 16;           // scan chunk (timesteps staged per TDM transfer)
constexpr int NCH  = T / CH;

// ---------------- types ----------------
typedef float    v8f    __attribute__((ext_vector_type(8)));
typedef __bf16   v16bf  __attribute__((ext_vector_type(16)));
typedef uint32_t u32x4  __attribute__((ext_vector_type(4)));
typedef int      i32x4v __attribute__((ext_vector_type(4)));
typedef int      i32x8v __attribute__((ext_vector_type(8)));

union Frag {
    uint4 q[2];
    v16bf v;
};

// ---------------- bf16 helpers ----------------
__device__ __forceinline__ uint16_t f32_to_bf16(float f) {
    uint32_t u = __builtin_bit_cast(uint32_t, f);
    uint32_t r = u + 0x7FFFu + ((u >> 16) & 1u);   // round-to-nearest-even
    return (uint16_t)(r >> 16);
}
__device__ __forceinline__ float bf16_to_f32(uint16_t h) {
    uint32_t u = ((uint32_t)h) << 16;
    return __builtin_bit_cast(float, u);
}

// ---------------- weight transpose + f32->bf16 convert ----------------
__global__ void tconv_kernel(uint16_t* __restrict__ dst, const float* __restrict__ src,
                             int K, int Nsrc, int Npad) {
    int k = blockIdx.x * 16 + threadIdx.x;
    int n = blockIdx.y * 16 + threadIdx.y;
    if (k < K && n < Npad) {
        dst[(size_t)n * K + k] = (n < Nsrc) ? f32_to_bf16(src[(size_t)k * Nsrc + n])
                                            : (uint16_t)0;
    }
}

// ---------------- token shift: sx (f32) and xxx (bf16) ----------------
__global__ void shift_mix_kernel(const float* __restrict__ x,
                                 const float* __restrict__ state,
                                 const uint8_t* __restrict__ ns,
                                 const float* __restrict__ tmx,
                                 float* __restrict__ sx,
                                 uint16_t* __restrict__ xxx) {
    int idx = blockIdx.x * blockDim.x + threadIdx.x;
    if (idx >= T * C) return;
    int t = idx >> 11;
    int c = idx & (C - 1);
    float xv = x[idx];
    float prev = (t == 0) ? state[c] : x[idx - C];
    if (ns[t]) prev = 0.0f;
    float sxv = prev - xv;
    sx[idx]  = sxv;
    xxx[idx] = f32_to_bf16(xv + sxv * tmx[c]);
}

// ---------------- WMMA GEMM ----------------
// out[M,N] = A[M,K](bf16) x Bt[N,K](bf16)^T
// 256 threads = 8 waves as 4(M) x 2(N); block tile 128x128; wave tile 32x64.
// K is compile-time; pipeline uses two statically-named fragment sets so
// everything stays in VGPRs (no dynamic array indexing -> no scratch demotion).
__device__ __forceinline__ void ldfrag(Frag& f, const uint16_t* p) {
    f.q[0] = *(const uint4*)p;
    f.q[1] = *(const uint4*)(p + 16);
}

__device__ __forceinline__ void load_chunk(Frag (&a)[2], Frag (&b)[4],
                                           const uint16_t* const (&arow)[2],
                                           const uint16_t* const (&brow)[4], int kk) {
#pragma unroll
    for (int mi = 0; mi < 2; ++mi) ldfrag(a[mi], arow[mi] + kk);
#pragma unroll
    for (int i = 0; i < 4; ++i) ldfrag(b[i], brow[i] + kk);
}

__device__ __forceinline__ void do_wmma(v8f (&acc)[2][4],
                                        const Frag (&a)[2], const Frag (&b)[4]) {
#pragma unroll
    for (int mi = 0; mi < 2; ++mi)
#pragma unroll
        for (int i = 0; i < 4; ++i)
            acc[mi][i] = __builtin_amdgcn_wmma_f32_16x16x32_bf16(
                false, a[mi].v, false, b[i].v, (short)0, acc[mi][i], false, false);
}

template <int EPI, int KT>
__global__ __launch_bounds__(256) void gemm_wmma_k(
    const uint16_t* __restrict__ A, int lda,
    const uint16_t* __restrict__ Bt, int ldb,
    int N,
    float* __restrict__ outF, uint16_t* __restrict__ outH, int ldo,
    const float* __restrict__ ex, const float* __restrict__ esx,
    const float* __restrict__ evec) {

    const int lane  = threadIdx.x & 31;
    const int wave  = threadIdx.x >> 5;
    const int wm    = wave >> 1;
    const int wn    = wave & 1;
    const int mBase = blockIdx.y * 128 + wm * 32;
    const int nBase = blockIdx.x * 128 + wn * 64;
    const int lrow  = lane & 15;
    const int kb    = (lane >> 4) << 3;

    const uint16_t* arow[2];
    arow[0] = A + (size_t)(mBase + lrow) * lda + kb;
    arow[1] = arow[0] + (size_t)16 * lda;
    const uint16_t* brow[4];
#pragma unroll
    for (int i = 0; i < 4; ++i)
        brow[i] = Bt + (size_t)(nBase + i * 16 + lrow) * ldb + kb;

    v8f zero = {0.f, 0.f, 0.f, 0.f, 0.f, 0.f, 0.f, 0.f};
    v8f acc[2][4];
#pragma unroll
    for (int mi = 0; mi < 2; ++mi)
#pragma unroll
        for (int i = 0; i < 4; ++i) acc[mi][i] = zero;

    constexpr int NK = KT / 32;            // number of 32-wide K chunks
    Frag aA[2], bA[4], aB[2], bB[4];
    load_chunk(aA, bA, arow, brow, 0);

#pragma unroll 1
    for (int it = 0; it + 1 < NK; it += 2) {
        __builtin_prefetch(arow[0] + it * 32 + 256, 0, 3);   // global_prefetch_b8
        load_chunk(aB, bB, arow, brow, (it + 1) * 32);
        do_wmma(acc, aA, bA);
        if (it + 2 < NK) load_chunk(aA, bA, arow, brow, (it + 2) * 32);
        do_wmma(acc, aB, bB);
    }
    if constexpr (NK & 1) do_wmma(acc, aA, bA);   // odd tail (covers NK==1)

    const int nc0 = nBase + lrow;
#pragma unroll
    for (int mi = 0; mi < 2; ++mi) {
        const int mrow = mBase + mi * 16 + ((lane >> 4) << 3);
#pragma unroll
        for (int i = 0; i < 4; ++i) {
            int n = nc0 + i * 16;
            if (n >= N) continue;
#pragma unroll
            for (int j = 0; j < 8; ++j) {
                size_t idx = (size_t)(mrow + j) * ldo + n;
                float val = acc[mi][i][j];
                if constexpr (EPI == 0) {
                    outF[idx] = val;
                } else if constexpr (EPI == 1) {
                    outH[idx] = f32_to_bf16(tanhf(val));
                } else if constexpr (EPI == 2) {
                    outH[idx] = f32_to_bf16(1.0f / (1.0f + __expf(-val)));
                } else if constexpr (EPI == 3) {
                    outH[idx] = f32_to_bf16(ex[idx] + esx[idx] * (evec[n] + val));
                } else if constexpr (EPI == 4) {
                    outF[idx] = fmaxf(-__expf(evec[n] + val), -5.0f);
                }
            }
        }
    }
}

// ---------------- TDM: 2D strided tile (rows x 64 f32) -> LDS ----------------
// D# per CDNA5 ISA §8; 6-arg builtin on this toolchain: (g0, g1, g2, g3, g4, cpol).
__device__ __forceinline__ void tdm_load_tile(uint32_t lds_off, const void* gptr,
                                              uint32_t stride_elems) {
    uint64_t ga = (uint64_t)(uintptr_t)gptr;
    u32x4 g0;
    g0[0] = 1u;                                    // count=1, user mode
    g0[1] = lds_off;                               // LDS byte address
    g0[2] = (uint32_t)ga;                          // global_addr[31:0]
    g0[3] = (uint32_t)(ga >> 32) | (2u << 30);     // global_addr[56:32] | type=2

    const uint32_t td0 = 1u << 20, td1 = 1u << 20; // oversize: no OOB clamping
    const uint32_t tile0 = 64, tile1 = CH, tile2 = 0;
    const uint64_t s0 = stride_elems, s1 = 0;
    i32x8v g1;
    g1[0] = (int)(2u << 16);                                       // data_size=4B
    g1[1] = (int)((td0 & 0xFFFFu) << 16);                          // tensor_dim0[15:0]
    g1[2] = (int)((td0 >> 16) | ((td1 & 0xFFFFu) << 16));          // td0[31:16]|td1[15:0]
    g1[3] = (int)((td1 >> 16) | (tile0 << 16));                    // td1[31:16]|tile_dim0
    g1[4] = (int)(tile1 | (tile2 << 16));                          // tile_dim1|tile_dim2
    g1[5] = (int)(uint32_t)s0;                                     // stride0[31:0]
    g1[6] = (int)(uint32_t)((s0 >> 32) | ((s1 & 0xFFFFu) << 16));  // stride0[47:32]|s1[15:0]
    g1[7] = (int)(uint32_t)(s1 >> 16);                             // stride1[47:16]
    i32x4v z4 = {0, 0, 0, 0};
    i32x8v z8 = {0, 0, 0, 0, 0, 0, 0, 0};
    __builtin_amdgcn_tensor_load_to_lds(g0, g1, z4, z4, z8, 0);
}

__device__ __forceinline__ uint32_t lds_off_of(const void* p) {
    return (uint32_t)(uintptr_t)p;   // low 32 bits of generic LDS pointer = LDS offset
}

// ---------------- WKV recurrence: one block per head ----------------
__global__ __launch_bounds__(256) void scan_kernel(
    const float* __restrict__ r,       // [T, H*S]
    const float* __restrict__ ksm,     // [T, HKV*S]
    const float* __restrict__ vsm,     // [T, HKV*S]
    const float* __restrict__ lw,      // [T, H*S]
    const uint8_t* __restrict__ ns,    // [T]
    const float* __restrict__ state_in,
    float* __restrict__ attout,        // [T, H*S]
    float* __restrict__ state_out) {

    const int h    = blockIdx.x;
    const int tid  = threadIdx.x;
    const int wave = tid >> 5;
    const int lane = tid & 31;
    const int v    = tid >> 2;
    const int kg   = tid & 3;
    const int k0   = kg << 4;

    __shared__ float l_r [2][CH][Ss];
    __shared__ float l_ew[2][CH][Ss];   // logw staged, converted to exp(lw) in pre-pass
    __shared__ float l_ke[2][CH][Ss];   // k staged, converted to k*(1-ew) in pre-pass
    __shared__ float l_v [2][CH][Ss];
    __shared__ int   l_ns[2][CH];

    float s[16];
    const float* sin = state_in + C + h * (Ss * Ss);
#pragma unroll
    for (int j = 0; j < 16; ++j) s[j] = sin[(k0 + j) * Ss + v];

    const int baseF0 = h * Ss;
    const int baseH0 = (h >> 2) * Ss;

    // prologue: stage chunk 0 into buffer 0
    if (wave == 0) {
        tdm_load_tile(lds_off_of(&l_r [0][0][0]), r   + baseF0, Hh * Ss);
        tdm_load_tile(lds_off_of(&l_ew[0][0][0]), lw  + baseF0, Hh * Ss);
        tdm_load_tile(lds_off_of(&l_ke[0][0][0]), ksm + baseH0, HKV * Ss);
        tdm_load_tile(lds_off_of(&l_v [0][0][0]), vsm + baseH0, HKV * Ss);
        if (lane < CH) l_ns[0][lane] = (int)ns[lane];
    }

    for (int ci = 0; ci < NCH; ++ci) {
        const int cur = ci & 1;
        // stage chunk ci+1 into the other buffer, then wait for chunk ci only
        if (wave == 0) {
            if (ci + 1 < NCH) {
                const int nb = cur ^ 1;
                const int t1 = (ci + 1) * CH;
                tdm_load_tile(lds_off_of(&l_r [nb][0][0]), r   + (size_t)t1 * (Hh * Ss)  + baseF0, Hh * Ss);
                tdm_load_tile(lds_off_of(&l_ew[nb][0][0]), lw  + (size_t)t1 * (Hh * Ss)  + baseF0, Hh * Ss);
                tdm_load_tile(lds_off_of(&l_ke[nb][0][0]), ksm + (size_t)t1 * (HKV * Ss) + baseH0, HKV * Ss);
                tdm_load_tile(lds_off_of(&l_v [nb][0][0]), vsm + (size_t)t1 * (HKV * Ss) + baseH0, HKV * Ss);
                if (lane < CH) l_ns[nb][lane] = (int)ns[t1 + lane];
                __builtin_amdgcn_s_wait_tensorcnt(4);
            } else {
                __builtin_amdgcn_s_wait_tensorcnt(0);
            }
        }
        __syncthreads();

        // pre-pass: l_ew = exp(logw); l_ke = k * (1 - ew)
#pragma unroll
        for (int p = 0; p < (CH * Ss) / 256; ++p) {
            int e  = tid + 256 * p;
            int st = e >> 6;
            int kk = e & 63;
            float ew = __expf(l_ew[cur][st][kk]);
            l_ew[cur][st][kk] = ew;
            l_ke[cur][st][kk] *= (1.0f - ew);
        }
        __syncthreads();

        // 16 sequential steps from LDS
        const int tB = ci * CH;
#pragma unroll 4
        for (int st = 0; st < CH; ++st) {
            const bool z  = (l_ns[cur][st] != 0);
            const float vt = l_v[cur][st][v];
            float acc = 0.0f;
#pragma unroll
            for (int j = 0; j < 16; ++j) {
                float sj = z ? 0.0f : s[j];
                sj = sj * l_ew[cur][st][k0 + j] + l_ke[cur][st][k0 + j] * vt;
                s[j] = sj;
                acc += l_r[cur][st][k0 + j] * sj;
            }
            acc += __shfl_xor(acc, 1, 32);
            acc += __shfl_xor(acc, 2, 32);
            if (kg == 0) attout[(size_t)(tB + st) * (Hh * Ss) + baseF0 + v] = acc;
        }
        __syncthreads();
    }

    float* so = state_out + C + h * (Ss * Ss);
#pragma unroll
    for (int j = 0; j < 16; ++j) so[(k0 + j) * Ss + v] = s[j];
}

// ---------------- y = attout * g  -> bf16 ----------------
__global__ void mul_g_kernel(const float* __restrict__ attout,
                             const uint16_t* __restrict__ g,
                             uint16_t* __restrict__ yin) {
    int idx = blockIdx.x * blockDim.x + threadIdx.x;
    if (idx >= T * C) return;
    yin[idx] = f32_to_bf16(attout[idx] * bf16_to_f32(g[idx]));
}

// ---------------- new_state row 0 = x[length-1] ----------------
__global__ void state0_kernel(const float* __restrict__ x,
                              const int* __restrict__ lenp,
                              float* __restrict__ state_out) {
    int c = blockIdx.x * blockDim.x + threadIdx.x;
    if (c >= C) return;
    int len = lenp[0];
    state_out[c] = x[(size_t)(len - 1) * C + c];
}

// ================= host-side orchestration =================
extern "C" void kernel_launch(void* const* d_in, const int* in_sizes, int n_in,
                              void* d_out, int out_size, void* d_ws, size_t ws_size,
                              hipStream_t stream) {
    const float*   x     = (const float*)d_in[0];
    const float*   state = (const float*)d_in[1];
    const uint8_t* ns    = (const uint8_t*)d_in[2];
    const int*     lenp  = (const int*)d_in[3];
    const float*   tmx   = (const float*)d_in[4];
    const float*   tmr   = (const float*)d_in[5];
    const float*   tmk   = (const float*)d_in[6];
    const float*   tmv   = (const float*)d_in[7];
    const float*   tmw   = (const float*)d_in[8];
    const float*   tmg   = (const float*)d_in[9];
    const float*   w1    = (const float*)d_in[10];
    const float*   w2    = (const float*)d_in[11];
    const float*   tdec  = (const float*)d_in[12];
    const float*   dw1   = (const float*)d_in[13];
    const float*   dw2   = (const float*)d_in[14];
    const float*   Wq    = (const float*)d_in[15];
    const float*   Wk    = (const float*)d_in[16];
    const float*   Wv    = (const float*)d_in[17];
    const float*   Wg    = (const float*)d_in[18];
    const float*   Wo    = (const float*)d_in[19];

    float* outY     = (float*)d_out;
    float* outState = (float*)d_out + (size_t)T * C;

    char* base = (char*)d_ws;
    size_t off = 0;
    auto alloc = [&](size_t bytes) -> char* {
        char* p = base + off;
        off = (off + bytes + 255) & ~(size_t)255;
        return p;
    };

    float*    sx     = (float*)   alloc((size_t)T * C * 4);   // reused as attout
    uint16_t* xxx    = (uint16_t*)alloc((size_t)T * C * 2);   // reused as yin
    uint16_t* mtanh  = (uint16_t*)alloc((size_t)T * 160 * 2);
    uint16_t* xb[5];
    for (int i = 0; i < 5; ++i) xb[i] = (uint16_t*)alloc((size_t)T * C * 2);
    float*    rbuf   = (float*)   alloc((size_t)T * (Hh * Ss) * 4);
    float*    kbuf   = (float*)   alloc((size_t)T * (HKV * Ss) * 4);
    float*    vbuf   = (float*)   alloc((size_t)T * (HKV * Ss) * 4);
    uint16_t* gbuf   = (uint16_t*)alloc((size_t)T * C * 2);
    uint16_t* t1buf  = (uint16_t*)alloc((size_t)T * LDEC * 2);
    float*    logw   = (float*)   alloc((size_t)T * C * 4);
    uint16_t* Wq_t   = (uint16_t*)alloc((size_t)C * C * 2);
    uint16_t* Wk_t   = (uint16_t*)alloc((size_t)(HKV * Ss) * C * 2);
    uint16_t* Wv_t   = (uint16_t*)alloc((size_t)(HKV * Ss) * C * 2);
    uint16_t* Wg_t   = (uint16_t*)alloc((size_t)C * C * 2);
    uint16_t* Wo_t   = (uint16_t*)alloc((size_t)C * C * 2);
    uint16_t* w1_t   = (uint16_t*)alloc((size_t)256 * C * 2);      // 160 padded to 256
    uint16_t* w2_t   = (uint16_t*)alloc((size_t)5 * C * LMAA * 2);
    uint16_t* dw1_t  = (uint16_t*)alloc((size_t)128 * C * 2);      // 64 padded to 128
    uint16_t* dw2_t  = (uint16_t*)alloc((size_t)C * LDEC * 2);

    float*    attout = sx;
    uint16_t* yin    = xxx;
    (void)ws_size; (void)in_sizes; (void)n_in; (void)out_size;

    dim3 tcb(16, 16);

    // ---- 1) weight convert + transpose ----
    tconv_kernel<<<dim3(C / 16, C / 16), tcb, 0, stream>>>(Wq_t, Wq, C, C, C);
    tconv_kernel<<<dim3(C / 16, (HKV * Ss) / 16), tcb, 0, stream>>>(Wk_t, Wk, C, HKV * Ss, HKV * Ss);
    tconv_kernel<<<dim3(C / 16, (HKV * Ss) / 16), tcb, 0, stream>>>(Wv_t, Wv, C, HKV * Ss, HKV * Ss);
    tconv_kernel<<<dim3(C / 16, C / 16), tcb, 0, stream>>>(Wg_t, Wg, C, C, C);
    tconv_kernel<<<dim3(C / 16, C / 16), tcb, 0, stream>>>(Wo_t, Wo, C, C, C);
    tconv_kernel<<<dim3(C / 16, 256 / 16), tcb, 0, stream>>>(w1_t, w1, C, 160, 256);
    tconv_kernel<<<dim3(C / 16, 128 / 16), tcb, 0, stream>>>(dw1_t, dw1, C, LDEC, 128);
    tconv_kernel<<<dim3(LDEC / 16, C / 16), tcb, 0, stream>>>(dw2_t, dw2, LDEC, C, C);
    for (int n = 0; n < 5; ++n) {
        tconv_kernel<<<dim3(LMAA / 16, C / 16), tcb, 0, stream>>>(
            w2_t + (size_t)n * C * LMAA, w2 + (size_t)n * LMAA * C, LMAA, C, C);
    }

    // ---- 2) token shift ----
    shift_mix_kernel<<<(T * C) / 256, 256, 0, stream>>>(x, state, ns, tmx, sx, xxx);

    // ---- 3) LoRA GEMM 1: mtanh[T,160] = tanh(xxx @ w1) ----
    gemm_wmma_k<1, C><<<dim3(2, T / 128), 256, 0, stream>>>(
        xxx, C, w1_t, C, 160, nullptr, mtanh, 160, nullptr, nullptr, nullptr);

    // ---- 4) five mixed branches ----
    const float* maav[5] = {tmr, tmk, tmv, tmw, tmg};
    for (int n = 0; n < 5; ++n) {
        gemm_wmma_k<3, LMAA><<<dim3(C / 128, T / 128), 256, 0, stream>>>(
            mtanh + n * LMAA, 160, w2_t + (size_t)n * C * LMAA, LMAA,
            C, nullptr, xb[n], C, x, sx, maav[n]);
    }

    // ---- 5) projections ----
    gemm_wmma_k<0, C><<<dim3(C / 128, T / 128), 256, 0, stream>>>(
        xb[0], C, Wq_t, C, C, rbuf, nullptr, C, nullptr, nullptr, nullptr);
    gemm_wmma_k<0, C><<<dim3((HKV * Ss) / 128, T / 128), 256, 0, stream>>>(
        xb[1], C, Wk_t, C, HKV * Ss, kbuf, nullptr, HKV * Ss, nullptr, nullptr, nullptr);
    gemm_wmma_k<0, C><<<dim3((HKV * Ss) / 128, T / 128), 256, 0, stream>>>(
        xb[2], C, Wv_t, C, HKV * Ss, vbuf, nullptr, HKV * Ss, nullptr, nullptr, nullptr);
    gemm_wmma_k<2, C><<<dim3(C / 128, T / 128), 256, 0, stream>>>(
        xb[4], C, Wg_t, C, C, nullptr, gbuf, C, nullptr, nullptr, nullptr);
    gemm_wmma_k<1, C><<<dim3(1, T / 128), 256, 0, stream>>>(
        xb[3], C, dw1_t, C, LDEC, nullptr, t1buf, LDEC, nullptr, nullptr, nullptr);
    gemm_wmma_k<4, LDEC><<<dim3(C / 128, T / 128), 256, 0, stream>>>(
        t1buf, LDEC, dw2_t, LDEC, C, logw, nullptr, C, nullptr, nullptr, tdec);

    // ---- 6) WKV recurrence (TDM-staged, double-buffered) ----
    scan_kernel<<<Hh, 256, 0, stream>>>(rbuf, kbuf, vbuf, logw, ns, state, attout, outState);

    // ---- 7) y = attout * g -> bf16 ----
    mul_g_kernel<<<(T * C) / 256, 256, 0, stream>>>(attout, gbuf, yin);

    // ---- 8) final projection into d_out ----
    gemm_wmma_k<0, C><<<dim3(C / 128, T / 128), 256, 0, stream>>>(
        yin, C, Wo_t, C, C, outY, nullptr, C, nullptr, nullptr, nullptr);

    // ---- 9) new_state[0] = x[length-1] ----
    state0_kernel<<<(C + 255) / 256, 256, 0, stream>>>(x, lenp, outState);
}
